// CEBlock_55336358642142
// MI455X (gfx1250) — matrix operand order repaired and checked by reading, not compiled, not run on hardware
//
#include <hip/hip_runtime.h>

// ---------------- problem constants ----------------
#define B_    32
#define LT    64
#define LS    256
#define LTOT  320
#define DIM_  768
#define H_    12
#define HD    64
#define HID   3072
#define KEEP  180            // ceil(0.7 * 256)
#define LNEW  244            // 64 + 180
#define QKVC  2304           // 3*DIM
#define SCALE 0.125f         // 64^-0.5

typedef __attribute__((ext_vector_type(16))) __bf16 v16bf;
typedef __attribute__((ext_vector_type(8)))  float  v8f;

union Frag { v16bf v; unsigned int u[8]; unsigned short s[16]; };

// ---------------- bf16 helpers (RNE) ----------------
__device__ __forceinline__ unsigned short f2bf(float f) {
  unsigned int u = __float_as_uint(f);
  u = (u + 0x7fffu + ((u >> 16) & 1u)) >> 16;
  return (unsigned short)u;
}

// A-matrix 16-bit layout: lane<16 -> K {0..7,16..23}, lane>=16 -> +8
__device__ __forceinline__ int kmapA(int i, int half) {
  return ((i < 8) ? i : (i + 8)) + half * 8;
}

// load A fragment from a bf16 row (4B aligned, k0 multiple of 32)
__device__ __forceinline__ void load_fragA(Frag& f, const unsigned short* row, int k0, int half) {
  const unsigned int* p = (const unsigned int*)row;
  int b0 = (k0 + half * 8) >> 1;
  int b1 = (k0 + 16 + half * 8) >> 1;
#pragma unroll
  for (int i = 0; i < 4; ++i) { f.u[i] = p[b0 + i]; f.u[4 + i] = p[b1 + i]; }
}

// B-matrix 16-bit layout: lane<16 -> K {0..7,16..23}, lane>=16 -> +8 (same as A)
__device__ __forceinline__ void load_fragB(Frag& f, const unsigned short* row, int k0, int half) {
  const unsigned int* p = (const unsigned int*)row;
  int b0 = (k0 + half * 8) >> 1;
  int b1 = (k0 + 16 + half * 8) >> 1;
#pragma unroll
  for (int i = 0; i < 4; ++i) { f.u[i] = p[b0 + i]; f.u[4 + i] = p[b1 + i]; }
}

// ---------------- elementwise f32 -> bf16 ----------------
__global__ void cvt_bf16(const float* __restrict__ in, unsigned short* __restrict__ out, int n) {
  int i = blockIdx.x * blockDim.x + threadIdx.x;
  if (i < n) out[i] = f2bf(in[i]);
}

// ---------------- layernorm (one wave per token) ----------------
__global__ void layernorm_bf16(const float* __restrict__ xin, const float* __restrict__ w,
                               const float* __restrict__ bb, unsigned short* __restrict__ out,
                               int ntok) {
  int tok  = blockIdx.x * (blockDim.x >> 5) + (threadIdx.x >> 5);
  int lane = threadIdx.x & 31;
  if (tok >= ntok) return;
  const float* row = xin + (size_t)tok * DIM_;
  float vals[DIM_ / 32];
  float s = 0.f;
#pragma unroll
  for (int i = 0; i < DIM_ / 32; ++i) { vals[i] = row[lane + i * 32]; s += vals[i]; }
#pragma unroll
  for (int o = 16; o > 0; o >>= 1) s += __shfl_xor(s, o, 32);
  float mu = s * (1.0f / DIM_);
  float vs = 0.f;
#pragma unroll
  for (int i = 0; i < DIM_ / 32; ++i) { float d = vals[i] - mu; vs += d * d; }
#pragma unroll
  for (int o = 16; o > 0; o >>= 1) vs += __shfl_xor(vs, o, 32);
  float inv = rsqrtf(vs * (1.0f / DIM_) + 1e-5f);
#pragma unroll
  for (int i = 0; i < DIM_ / 32; ++i) {
    int c = lane + i * 32;
    out[(size_t)tok * DIM_ + c] = f2bf((vals[i] - mu) * inv * w[c] + bb[c]);
  }
}

// ---------------- generic WMMA GEMM with async A-tile staging ----------------
// C = act(A * W^T + bias (+resid)); A:[M,K] bf16, W:[N,K] bf16.
// Block = 4 waves sharing one 16(M) x 64(K) A-tile, double-buffered in LDS,
// filled by global_load_async_to_lds_b128 and drained under s_wait_asynccnt.
// Requires: M%16==0, N%64==0, K%64==0 (true for all call sites).
template<int GELU, int OUTBF16>
__global__ void gemm_wmma(const unsigned short* __restrict__ A,
                          const unsigned short* __restrict__ W,
                          const float* __restrict__ bias,
                          const float* __restrict__ resid,
                          void* __restrict__ outp, int M, int N, int K) {
  __shared__ __align__(16) unsigned short atile[2][16 * 64];  // 2 x 2KB
  int tid  = threadIdx.x;
  int wave = tid >> 5;
  int lane = tid & 31;
  int half = lane >> 4;
  int lr   = lane & 15;
  int m0 = blockIdx.x * 16;
  int n0 = (blockIdx.y * 4 + wave) * 16;

  // copy mapping: 128 lanes x 16B cover the 16x64 bf16 tile
  int crow = tid >> 3;          // 0..15 (tile row)
  int ccol = (tid & 7) * 8;     // element offset in row (8 ushorts = 16B)
  unsigned long long gbase =
      (unsigned long long)(A + (size_t)(m0 + crow) * K + ccol);

  // prefetch k-tile 0 into buffer 0
  {
    unsigned lds = (unsigned)(size_t)&atile[0][crow * 64 + ccol];
    asm volatile("global_load_async_to_lds_b128 %0, %1, off"
                 :: "v"(lds), "v"(gbase) : "memory");
  }

  const unsigned short* wrow = W + (size_t)(n0 + lr) * K;
  v8f acc = {};
  int nk = K >> 6;
  for (int it = 0; it < nk; ++it) {
    asm volatile("s_wait_asynccnt 0x0" ::: "memory");  // my copies of tile `it` done
    __syncthreads();                                   // everyone's copies done; prev reads done
    if (it + 1 < nk) {
      unsigned lds = (unsigned)(size_t)&atile[(it + 1) & 1][crow * 64 + ccol];
      unsigned long long ga = gbase + (unsigned long long)((it + 1) << 6) * 2ull;
      asm volatile("global_load_async_to_lds_b128 %0, %1, off"
                   :: "v"(lds), "v"(ga) : "memory");
    }
    const unsigned short* arow = &atile[it & 1][lr * 64];
    int kb = it << 6;
#pragma unroll
    for (int kk = 0; kk < 64; kk += 32) {
      __builtin_prefetch((const void*)(wrow + kb + kk + 128), 0, 1);
      Frag fa, fb;
      load_fragA(fa, arow, kk, half);        // from LDS (ds_load)
      load_fragB(fb, wrow, kb + kk, half);   // from global
      acc = __builtin_amdgcn_wmma_f32_16x16x32_bf16(false, fa.v, false, fb.v,
                                                    (short)0, acc, false, false);
    }
  }

  int n = n0 + lr;
  float bv = bias ? bias[n] : 0.f;
#pragma unroll
  for (int r = 0; r < 8; ++r) {
    int m = m0 + r + half * 8;
    float v = acc[r] + bv;
    if (resid) v += resid[(size_t)m * N + n];
    if (GELU)  v = 0.5f * v * (1.f + erff(v * 0.70710678f));
    size_t o = (size_t)m * N + n;
    if (OUTBF16) ((unsigned short*)outp)[o] = f2bf(v);
    else         ((float*)outp)[o]          = v;
  }
}

// ---------------- attention scores + softmax -> attn (f32, to d_out) ----------------
__global__ void attn_softmax(const unsigned short* __restrict__ qkv,
                             float* __restrict__ attn_out) {
  __shared__ float s[16][LTOT];
  int bh = blockIdx.x, b = bh / H_, h = bh % H_;
  int l0 = blockIdx.y * 16;
  int wave = threadIdx.x >> 5, lane = threadIdx.x & 31, half = lane >> 4, lr = lane & 15;
  const unsigned short* qrow = qkv + (size_t)(b * LTOT + l0 + lr) * QKVC + h * HD;
  for (int t = wave; t < LTOT / 16; t += 4) {
    int m0 = t * 16;
    const unsigned short* krow = qkv + (size_t)(b * LTOT + m0 + lr) * QKVC + DIM_ + h * HD;
    v8f acc = {};
#pragma unroll
    for (int k0 = 0; k0 < HD; k0 += 32) {
      Frag fa, fb;
      load_fragA(fa, qrow, k0, half);
      load_fragB(fb, krow, k0, half);
      acc = __builtin_amdgcn_wmma_f32_16x16x32_bf16(false, fa.v, false, fb.v,
                                                    (short)0, acc, false, false);
    }
#pragma unroll
    for (int r = 0; r < 8; ++r) s[r + half * 8][m0 + lr] = acc[r] * SCALE;
  }
  __syncthreads();
  int row = threadIdx.x >> 3;
  int sub = threadIdx.x & 7;
  float mx = -1e30f;
  for (int j = sub; j < LTOT; j += 8) mx = fmaxf(mx, s[row][j]);
#pragma unroll
  for (int o = 4; o > 0; o >>= 1) mx = fmaxf(mx, __shfl_xor(mx, o, 8));
  float sum = 0.f;
  for (int j = sub; j < LTOT; j += 8) { float e = __expf(s[row][j] - mx); s[row][j] = e; sum += e; }
#pragma unroll
  for (int o = 4; o > 0; o >>= 1) sum += __shfl_xor(sum, o, 8);
  float inv = 1.f / sum;
  float* dst = attn_out + ((size_t)bh * LTOT + l0) * LTOT;
  for (int j = sub; j < LTOT; j += 8) dst[(size_t)row * LTOT + j] = s[row][j] * inv;
}

// ---------------- x_attn = attn @ V (bf16 out, [B,L,DIM] head-interleaved) ----------------
__global__ void attn_v(const float* __restrict__ attn, const unsigned short* __restrict__ qkv,
                       unsigned short* __restrict__ xattn) {
  int bh = blockIdx.x, b = bh / H_, h = bh % H_;
  int l0 = blockIdx.y * 16;
  int wave = threadIdx.x >> 5, lane = threadIdx.x & 31, half = lane >> 4, lr = lane & 15;
  int d0 = wave * 16;
  const float* arow = attn + ((size_t)bh * LTOT + l0 + lr) * LTOT;
  v8f acc = {};
  for (int k0 = 0; k0 < LTOT; k0 += 32) {
    Frag fa, fb;
#pragma unroll
    for (int i = 0; i < 16; ++i) fa.s[i] = f2bf(arow[k0 + kmapA(i, half)]);
#pragma unroll
    for (int i = 0; i < 16; ++i) {
      int m = k0 + kmapA(i, half);
      fb.s[i] = qkv[(size_t)(b * LTOT + m) * QKVC + 2 * DIM_ + h * HD + d0 + lr];
    }
    acc = __builtin_amdgcn_wmma_f32_16x16x32_bf16(false, fa.v, false, fb.v,
                                                  (short)0, acc, false, false);
  }
#pragma unroll
  for (int r = 0; r < 8; ++r) {
    int l = l0 + r + half * 8;
    xattn[(size_t)(b * LTOT + l) * DIM_ + h * HD + d0 + lr] = f2bf(acc[r]);
  }
}

// ---------------- attn_t[b,s] = mean_{h,l<64} attn[b,h,l,64+s] ----------------
__global__ void attn_pool(const float* __restrict__ attn, float* __restrict__ scores) {
  int b = blockIdx.x, s = threadIdx.x;  // 256 threads
  float acc = 0.f;
  for (int h = 0; h < H_; ++h)
    for (int l = 0; l < LT; ++l)
      acc += attn[(((size_t)(b * H_ + h)) * LTOT + l) * LTOT + LT + s];
  scores[b * LS + s] = acc * (1.f / (H_ * LT));
}

// ---------------- stable descending bitonic sort + index outputs ----------------
__global__ void sort_select(const float* __restrict__ scores, const int* __restrict__ gis,
                            const int* __restrict__ git, int* __restrict__ topk,
                            float* __restrict__ out_git, float* __restrict__ out_keep,
                            float* __restrict__ out_rem) {
  __shared__ float sv[LS];
  __shared__ int   si[LS];
  int b = blockIdx.x, t = threadIdx.x;
  sv[t] = scores[b * LS + t]; si[t] = t;
  __syncthreads();
  for (int k = 2; k <= LS; k <<= 1) {
    for (int j = k >> 1; j > 0; j >>= 1) {
      int ixj = t ^ j;
      if (ixj > t) {
        float va = sv[t], vb = sv[ixj];
        int   ia = si[t], ib = si[ixj];
        bool a_first = (va > vb) || (va == vb && ia < ib);  // key = (-score, idx)
        bool up = ((t & k) == 0);
        if (up ? (!a_first) : a_first) { sv[t] = vb; sv[ixj] = va; si[t] = ib; si[ixj] = ia; }
      }
      __syncthreads();
    }
  }
  if (t < KEEP) {
    topk[b * KEEP + t] = si[t];
    out_keep[b * KEEP + t] = (float)gis[b * LS + si[t]];
  } else {
    out_rem[b * (LS - KEEP) + (t - KEEP)] = (float)gis[b * LS + si[t]];
  }
  if (t < LT) out_git[b * LT + t] = (float)git[b * LT + t];
}

// ---------------- gather pruned tokens + LN2 ----------------
__global__ void gather_ln2(const float* __restrict__ x2, const int* __restrict__ topk,
                           const float* __restrict__ w, const float* __restrict__ bb,
                           float* __restrict__ xpruned, unsigned short* __restrict__ xn2) {
  int tok  = blockIdx.x * (blockDim.x >> 5) + (threadIdx.x >> 5);
  int lane = threadIdx.x & 31;
  if (tok >= B_ * LNEW) return;
  int b = tok / LNEW, l = tok % LNEW;
  int src = (l < LT) ? l : (LT + topk[b * KEEP + (l - LT)]);
  const float* row = x2 + (size_t)(b * LTOT + src) * DIM_;
  float vals[DIM_ / 32];
  float s = 0.f;
#pragma unroll
  for (int i = 0; i < DIM_ / 32; ++i) { vals[i] = row[lane + i * 32]; s += vals[i]; }
#pragma unroll
  for (int o = 16; o > 0; o >>= 1) s += __shfl_xor(s, o, 32);
  float mu = s * (1.0f / DIM_);
  float vs = 0.f;
#pragma unroll
  for (int i = 0; i < DIM_ / 32; ++i) { float d = vals[i] - mu; vs += d * d; }
#pragma unroll
  for (int o = 16; o > 0; o >>= 1) vs += __shfl_xor(vs, o, 32);
  float inv = rsqrtf(vs * (1.0f / DIM_) + 1e-5f);
#pragma unroll
  for (int i = 0; i < DIM_ / 32; ++i) {
    int c = lane + i * 32;
    xpruned[(size_t)tok * DIM_ + c] = vals[i];
    xn2[(size_t)tok * DIM_ + c] = f2bf((vals[i] - mu) * inv * w[c] + bb[c]);
  }
}

// ---------------- workspace layout (bytes) ----------------
static constexpr size_t OFF_WQKV   = 0;
static constexpr size_t OFF_WPROJ  = OFF_WQKV   + (size_t)QKVC * DIM_ * 2;
static constexpr size_t OFF_WFC1   = OFF_WPROJ  + (size_t)DIM_ * DIM_ * 2;
static constexpr size_t OFF_WFC2   = OFF_WFC1   + (size_t)HID  * DIM_ * 2;
static constexpr size_t OFF_XN     = OFF_WFC2   + (size_t)DIM_ * HID  * 2;
static constexpr size_t OFF_QKV    = OFF_XN     + (size_t)B_ * LTOT * DIM_ * 2;
static constexpr size_t OFF_XATTN  = OFF_QKV    + (size_t)B_ * LTOT * QKVC * 2;
static constexpr size_t OFF_X2     = OFF_XATTN  + (size_t)B_ * LTOT * DIM_ * 2;
static constexpr size_t OFF_SCORES = OFF_X2     + (size_t)B_ * LTOT * DIM_ * 4;
static constexpr size_t OFF_TOPK   = OFF_SCORES + (size_t)B_ * LS * 4;
static constexpr size_t OFF_XPR    = OFF_TOPK   + (size_t)B_ * KEEP * 4;
static constexpr size_t OFF_XN2    = OFF_XPR    + (size_t)B_ * LNEW * DIM_ * 4;
static constexpr size_t OFF_H      = OFF_XN2    + (size_t)B_ * LNEW * DIM_ * 2;

// d_out offsets (floats)
static constexpr size_t OUT_X    = 0;
static constexpr size_t OUT_GIT  = OUT_X    + (size_t)B_ * LNEW * DIM_;
static constexpr size_t OUT_KEEP = OUT_GIT  + (size_t)B_ * LT;
static constexpr size_t OUT_REM  = OUT_KEEP + (size_t)B_ * KEEP;
static constexpr size_t OUT_ATTN = OUT_REM  + (size_t)B_ * (LS - KEEP);

extern "C" void kernel_launch(void* const* d_in, const int* in_sizes, int n_in,
                              void* d_out, int out_size, void* d_ws, size_t ws_size,
                              hipStream_t stream) {
  (void)in_sizes; (void)n_in; (void)out_size; (void)ws_size;
  const float* x      = (const float*)d_in[0];
  const int*   git    = (const int*)d_in[1];
  const int*   gis    = (const int*)d_in[2];
  const float* n1w    = (const float*)d_in[3];
  const float* n1b    = (const float*)d_in[4];
  const float* qkv_w  = (const float*)d_in[5];
  const float* qkv_b  = (const float*)d_in[6];
  const float* proj_w = (const float*)d_in[7];
  const float* proj_b = (const float*)d_in[8];
  const float* n2w    = (const float*)d_in[9];
  const float* n2b    = (const float*)d_in[10];
  const float* fc1_w  = (const float*)d_in[11];
  const float* fc1_b  = (const float*)d_in[12];
  const float* fc2_w  = (const float*)d_in[13];
  const float* fc2_b  = (const float*)d_in[14];

  char* ws = (char*)d_ws;
  unsigned short* wqkv  = (unsigned short*)(ws + OFF_WQKV);
  unsigned short* wproj = (unsigned short*)(ws + OFF_WPROJ);
  unsigned short* wfc1  = (unsigned short*)(ws + OFF_WFC1);
  unsigned short* wfc2  = (unsigned short*)(ws + OFF_WFC2);
  unsigned short* xn    = (unsigned short*)(ws + OFF_XN);
  unsigned short* qkvb  = (unsigned short*)(ws + OFF_QKV);
  unsigned short* xattn = (unsigned short*)(ws + OFF_XATTN);
  float*          x2    = (float*)(ws + OFF_X2);
  float*          scrs  = (float*)(ws + OFF_SCORES);
  int*            topk  = (int*)(ws + OFF_TOPK);
  float*          xpr   = (float*)(ws + OFF_XPR);
  unsigned short* xn2   = (unsigned short*)(ws + OFF_XN2);
  unsigned short* hbuf  = (unsigned short*)(ws + OFF_H);

  float* out      = (float*)d_out;
  float* out_x    = out + OUT_X;
  float* out_git  = out + OUT_GIT;
  float* out_keep = out + OUT_KEEP;
  float* out_rem  = out + OUT_REM;
  float* out_attn = out + OUT_ATTN;

  // 1) weights -> bf16
  cvt_bf16<<<(QKVC * DIM_ + 255) / 256, 256, 0, stream>>>(qkv_w,  wqkv,  QKVC * DIM_);
  cvt_bf16<<<(DIM_ * DIM_ + 255) / 256, 256, 0, stream>>>(proj_w, wproj, DIM_ * DIM_);
  cvt_bf16<<<(HID  * DIM_ + 255) / 256, 256, 0, stream>>>(fc1_w,  wfc1,  HID  * DIM_);
  cvt_bf16<<<(DIM_ * HID  + 255) / 256, 256, 0, stream>>>(fc2_w,  wfc2,  DIM_ * HID);

  // 2) LN1
  layernorm_bf16<<<(B_ * LTOT) / 8, 256, 0, stream>>>(x, n1w, n1b, xn, B_ * LTOT);

  // 3) QKV GEMM
  gemm_wmma<0, 1><<<dim3((B_ * LTOT) / 16, QKVC / 64), 128, 0, stream>>>(
      xn, wqkv, qkv_b, nullptr, qkvb, B_ * LTOT, QKVC, DIM_);

  // 4) attention scores + softmax (attn -> d_out)
  attn_softmax<<<dim3(B_ * H_, LTOT / 16), 128, 0, stream>>>(qkvb, out_attn);

  // 5) attn @ V
  attn_v<<<dim3(B_ * H_, LTOT / 16), 128, 0, stream>>>(out_attn, qkvb, xattn);

  // 6) proj GEMM + residual(x) -> x2
  gemm_wmma<0, 0><<<dim3((B_ * LTOT) / 16, DIM_ / 64), 128, 0, stream>>>(
      xattn, wproj, proj_b, x, x2, B_ * LTOT, DIM_, DIM_);

  // 7) pruning scores, sort, index outputs
  attn_pool<<<B_, LS, 0, stream>>>(out_attn, scrs);
  sort_select<<<B_, LS, 0, stream>>>(scrs, gis, git, topk, out_git, out_keep, out_rem);

  // 8) gather pruned tokens + LN2
  gather_ln2<<<(B_ * LNEW) / 8, 256, 0, stream>>>(x2, topk, n2w, n2b, xpr, xn2);

  // 9) FC1 + GELU
  gemm_wmma<1, 1><<<dim3((B_ * LNEW) / 16, HID / 64), 128, 0, stream>>>(
      xn2, wfc1, fc1_b, nullptr, hbuf, B_ * LNEW, HID, DIM_);

  // 10) FC2 + residual(xpruned) -> d_out x
  gemm_wmma<0, 0><<<dim3((B_ * LNEW) / 16, DIM_ / 64), 128, 0, stream>>>(
      hbuf, wfc2, fc2_b, xpr, out_x, B_ * LNEW, DIM_, HID);
}
